// LayerNormLSTM_5669356836108
// MI455X (gfx1250) — matrix operand order repaired and checked by reading, not compiled, not run on hardware
//
#include <hip/hip_runtime.h>
#include <hip/hip_bf16.h>

// ---------------------------------------------------------------------------
// LayerNorm-LSTM for MI455X (gfx1250, wave32, WMMA). T=512, B=64, I=H=512.
//
// Phase 1: Wx = LN(x @ W, gamma0)        -- large bf16 WMMA GEMM, LN fused.
// Phase 2: ONE persistent kernel, 4 independent blocks (16 batch rows each,
//          batch rows are independent LSTM chains -> zero cross-block sync).
//          h (bf16) and c (fp32) live in LDS for all 512 steps; Wx[t] is
//          prefetched (global_prefetch_b8) under the GEMM to hide HBM latency.
// ---------------------------------------------------------------------------

#define TT   512
#define BB   64
#define II   512
#define HH   512
#define GG   2048          // 4*H
#define KD   512           // K of both GEMMs
#define NTW1 16            // phase-1: N-tiles per wave (8 waves * 16 * 16 = 2048)
#define NTW2 8             // phase-2: N-tiles per wave (16 waves * 8 * 16 = 2048)
#define HPAD 520           // LDS h row stride (ushorts): +8 kills bank conflicts
#define VSTR 2052          // LDS v row stride (floats): 8*(2052%64)=32 -> halves
                           // of a wave land 32 banks apart (no 2-way conflict)

typedef __attribute__((ext_vector_type(16))) __bf16 v16bf;
typedef __attribute__((ext_vector_type(8)))  float  v8f;

union ABFrag { uint4 u4[2]; v16bf v; };

__device__ __forceinline__ unsigned short f2bf(float f) {
  unsigned int u = __float_as_uint(f);
  unsigned int r = u + 0x7fffu + ((u >> 16) & 1u);   // round-to-nearest-even
  return (unsigned short)(r >> 16);
}

__device__ __forceinline__ float sigf(float x) { return 1.0f / (1.0f + expf(-x)); }

// -------------------------- elementwise converts ---------------------------

__global__ void f32_to_bf16_kernel(const float* __restrict__ src,
                                   unsigned short* __restrict__ dst, int n) {
  int i = blockIdx.x * blockDim.x + threadIdx.x;
  if (i < n) dst[i] = f2bf(src[i]);
}

// Pack row-major fp32 B[K][N] into WMMA bf16 B-fragment order:
// frag (kt,nt): lane l holds j=0..15 = B[kt*32 + (l>>4)*8 + j + (j>=8 ? 8 : 0)][nt*16 + (l&15)]
// -> each lane's 16 bf16 values contiguous (32B, one v16bf load).
__global__ void pack_b_kernel(const float* __restrict__ B,
                              unsigned short* __restrict__ Bp, int K, int N) {
  int idx = blockIdx.x * blockDim.x + threadIdx.x;   // one thread per (frag,lane)
  int NT = N / 16;
  int total = (K / 32) * NT * 32;
  if (idx >= total) return;
  int lane = idx & 31;
  int frag = idx >> 5;
  int nt = frag % NT;
  int kt = frag / NT;
  int col  = nt * 16 + (lane & 15);
  int krow = kt * 32 + (lane >> 4) * 8;
  unsigned short* out = Bp + (size_t)idx * 16;
#pragma unroll
  for (int j = 0; j < 16; ++j) out[j] = f2bf(B[(size_t)(krow + j + (j >= 8 ? 8 : 0)) * N + col]);
}

__global__ void init_state_kernel(const float* __restrict__ h0,
                                  const float* __restrict__ c0,
                                  float* __restrict__ hout0,
                                  float* __restrict__ cout0, int n) {
  int i = blockIdx.x * blockDim.x + threadIdx.x;
  if (i < n) { hout0[i] = h0[i]; cout0[i] = c0[i]; }
}

// --------------------- phase 1: GEMM + fused LayerNorm ---------------------
// One block = 16 rows x 2048 cols; 8 waves x 16 N-tiles each.
__global__ __launch_bounds__(256) void gemm_ln_kernel(
    const unsigned short* __restrict__ A,   // [Mtot][512] bf16 row-major
    const unsigned short* __restrict__ Bp,  // packed fragments
    const float* __restrict__ gamma_ln,     // [2048]
    float* __restrict__ Out)                // [Mtot][2048] fp32
{
  __shared__ float s_sum[16];
  __shared__ float s_sq[16];

  const int tid  = threadIdx.x;
  const int wave = tid >> 5;
  const int lane = tid & 31;
  const int half = lane >> 4;
  const int lrow = lane & 15;
  const int m0   = blockIdx.x * 16;
  const int nt0  = wave * NTW1;

  if (tid < 16) { s_sum[tid] = 0.0f; s_sq[tid] = 0.0f; }
  __syncthreads();

  v8f acc[NTW1];
  const v8f vzero = {0.f, 0.f, 0.f, 0.f, 0.f, 0.f, 0.f, 0.f};
#pragma unroll
  for (int i = 0; i < NTW1; ++i) acc[i] = vzero;

  const unsigned short* arow = A + (size_t)(m0 + lrow) * KD + half * 8;

  for (int kt = 0; kt < KD / 32; ++kt) {
    ABFrag af;
    af.u4[0] = *(const uint4*)(arow + kt * 32);
    af.u4[1] = *(const uint4*)(arow + kt * 32 + 16);
    const unsigned short* bbase =
        Bp + (((size_t)kt * (GG / 16) + nt0) * 32 + lane) * 16;
#pragma unroll
    for (int nt = 0; nt < NTW1; ++nt) {
      v16bf bf = *(const v16bf*)(bbase + (size_t)nt * (32 * 16));
      acc[nt] = __builtin_amdgcn_wmma_f32_16x16x32_bf16(
          false, af.v, false, bf, (short)0, acc[nt], false, false);
    }
  }

  float psum[8], psq[8];
#pragma unroll
  for (int v = 0; v < 8; ++v) { psum[v] = 0.0f; psq[v] = 0.0f; }
#pragma unroll
  for (int nt = 0; nt < NTW1; ++nt)
#pragma unroll
    for (int v = 0; v < 8; ++v) {
      float x = acc[nt][v];
      psum[v] += x;
      psq[v]  += x * x;
    }
  // 16-lane pre-reduction (xor<16 keeps lane halves separate), then 2 atomics.
#pragma unroll
  for (int v = 0; v < 8; ++v) {
#pragma unroll
    for (int off = 1; off < 16; off <<= 1) {
      psum[v] += __shfl_xor(psum[v], off, 32);
      psq[v]  += __shfl_xor(psq[v],  off, 32);
    }
  }
  if (lrow == 0) {
#pragma unroll
    for (int v = 0; v < 8; ++v) {
      atomicAdd(&s_sum[v + half * 8], psum[v]);   // ds_add_f32
      atomicAdd(&s_sq [v + half * 8], psq[v]);
    }
  }
  __syncthreads();

  float mu_[8], rstd_[8];
#pragma unroll
  for (int v = 0; v < 8; ++v) {
    int r = v + half * 8;
    float mu  = s_sum[r] * (1.0f / GG);
    float var = s_sq[r] * (1.0f / GG) - mu * mu;
    mu_[v]   = mu;
    rstd_[v] = rsqrtf(var + 1e-5f);
  }

#pragma unroll
  for (int nt = 0; nt < NTW1; ++nt) {
    int col = (nt0 + nt) * 16 + lrow;
    float gm = gamma_ln[col];
#pragma unroll
    for (int v = 0; v < 8; ++v) {
      int r = v + half * 8;
      Out[(size_t)(m0 + r) * GG + col] = (acc[nt][v] - mu_[v]) * rstd_[v] * gm;
    }
  }
}

// ------------------- phase 2: persistent fused recurrence ------------------
// grid = 4 blocks (16 batch rows each), 512 threads = 16 waves.
// LDS (dynamic): v_sh[16][2052] f32 | c_sh[16][512] f32 | h_sh[16][520] bf16.
// Total ~177KB < 320KB/WGP.
__global__ __launch_bounds__(512) void lstm_persistent_kernel(
    const unsigned short* __restrict__ Rp,    // packed recurrent weights
    const float* __restrict__ Wx,             // [T][B][2048] (post-LN)
    const float* __restrict__ bias,           // [2048]
    const float* __restrict__ gamma1,         // [2048]
    const float* __restrict__ gamma_h,        // [512]
    const float* __restrict__ beta_h,         // [512]
    const float* __restrict__ h0,             // [B][512]
    const float* __restrict__ c0,             // [B][512]
    float* __restrict__ h_base,               // [T+1][B][512]
    float* __restrict__ c_base)               // [T+1][B][512]
{
  extern __shared__ float smem[];
  float*          v_sh = smem;                           // 16*VSTR f32
  float*          c_sh = smem + 16 * VSTR;               // 16*512  f32
  unsigned short* h_sh = (unsigned short*)(c_sh + 16 * HH);  // 16*HPAD bf16

  __shared__ float s_sum[16], s_sq[16], s_cs[16], s_cq[16];

  const int tid  = threadIdx.x;        // 0..511
  const int wave = tid >> 5;           // 0..15
  const int lane = tid & 31;
  const int half = lane >> 4;
  const int lrow = lane & 15;
  const int b0   = blockIdx.x * 16;
  const int nt0  = wave * NTW2;
  const int hh   = tid;                // hidden index for cell math

  // Hoist all t-invariant loads into registers.
  float gm_[NTW2], bv_[NTW2];
#pragma unroll
  for (int nt = 0; nt < NTW2; ++nt) {
    int col = (nt0 + nt) * 16 + lrow;
    gm_[nt] = gamma1[col];
    bv_[nt] = bias[col];
  }
  const float gmh = gamma_h[hh];
  const float bth = beta_h[hh];

  // Load resident state: h -> LDS bf16 (padded rows), c -> LDS f32.
  for (int r = 0; r < 16; ++r) {
    c_sh[r * HH + hh]   = c0[(size_t)(b0 + r) * HH + hh];
    h_sh[r * HPAD + hh] = f2bf(h0[(size_t)(b0 + r) * HH + hh]);
  }

  const v8f vzero = {0.f, 0.f, 0.f, 0.f, 0.f, 0.f, 0.f, 0.f};

  for (int t = 0; t < TT; ++t) {
    // This step's Wx tile (128KB/block): prefetch NOW so the whole GEMM hides
    // the HBM fetch; the epilogue loads below then hit WGP$/L2.
    const float* wxt = Wx + ((size_t)t * BB + b0) * GG;
    __builtin_prefetch(wxt + (size_t)tid * 64, 0, 3);        // global_prefetch_b8
    __builtin_prefetch(wxt + (size_t)tid * 64 + 32, 0, 3);   // (2x128B per thread)

    if (tid < 16) { s_sum[tid] = 0.f; s_sq[tid] = 0.f; s_cs[tid] = 0.f; s_cq[tid] = 0.f; }
    __syncthreads();   // also orders prev-step h_sh writes before ds reads

    // ---- GEMM: v_raw = h @ R, A-fragments from LDS, B from L2 ----
    v8f acc[NTW2];
#pragma unroll
    for (int i = 0; i < NTW2; ++i) acc[i] = vzero;

    const unsigned short* arow = h_sh + lrow * HPAD + half * 8;
    for (int kt = 0; kt < KD / 32; ++kt) {
      ABFrag af;
      af.u4[0] = *(const uint4*)(arow + kt * 32);        // ds_read_b128
      af.u4[1] = *(const uint4*)(arow + kt * 32 + 16);
      const unsigned short* bbase =
          Rp + (((size_t)kt * (GG / 16) + nt0) * 32 + lane) * 16;
#pragma unroll
      for (int nt = 0; nt < NTW2; ++nt) {
        v16bf bf = *(const v16bf*)(bbase + (size_t)nt * (32 * 16));
        acc[nt] = __builtin_amdgcn_wmma_f32_16x16x32_bf16(
            false, af.v, false, bf, (short)0, acc[nt], false, false);
      }
    }

    // ---- LN statistics over the 2048-wide row ----
    float psum[8], psq[8];
#pragma unroll
    for (int v = 0; v < 8; ++v) { psum[v] = 0.0f; psq[v] = 0.0f; }
#pragma unroll
    for (int nt = 0; nt < NTW2; ++nt)
#pragma unroll
      for (int v = 0; v < 8; ++v) {
        float x = acc[nt][v];
        psum[v] += x;
        psq[v]  += x * x;
      }
#pragma unroll
    for (int v = 0; v < 8; ++v) {
#pragma unroll
      for (int off = 1; off < 16; off <<= 1) {
        psum[v] += __shfl_xor(psum[v], off, 32);
        psq[v]  += __shfl_xor(psq[v],  off, 32);
      }
    }
    if (lrow == 0) {
#pragma unroll
      for (int v = 0; v < 8; ++v) {
        atomicAdd(&s_sum[v + half * 8], psum[v]);
        atomicAdd(&s_sq [v + half * 8], psq[v]);
      }
    }
    __syncthreads();

    // ---- v = LN(raw)*gamma1 + Wx[t] + bias  -> v_sh ----
    float mu_[8], rstd_[8];
#pragma unroll
    for (int v = 0; v < 8; ++v) {
      int r = v + half * 8;
      float mu  = s_sum[r] * (1.0f / GG);
      float var = s_sq[r] * (1.0f / GG) - mu * mu;
      mu_[v]   = mu;
      rstd_[v] = rsqrtf(var + 1e-5f);
    }
#pragma unroll
    for (int nt = 0; nt < NTW2; ++nt) {
      int col = (nt0 + nt) * 16 + lrow;
#pragma unroll
      for (int v = 0; v < 8; ++v) {
        int r = v + half * 8;
        v_sh[r * VSTR + col] =
            (acc[nt][v] - mu_[v]) * rstd_[v] * gm_[nt] + wxt[(size_t)r * GG + col] + bv_[nt];
      }
    }
    __syncthreads();

    // ---- cell: gates + c_new (thread hh handles all 16 rows) ----
    float cn[16];
#pragma unroll
    for (int r = 0; r < 16; ++r) {
      float iv = v_sh[r * VSTR + hh];
      float gv = v_sh[r * VSTR + hh + 512];
      float fv = v_sh[r * VSTR + hh + 1024];
      float cp = c_sh[r * HH + hh];
      cn[r] = sigf(fv) * cp + sigf(iv) * tanhf(gv);
      c_sh[r * HH + hh] = cn[r];
    }
    __syncthreads();

    // ---- LN(c_new): one wave reduces one row (16 waves, 16 rows) ----
    {
      float s = 0.f, q = 0.f;
#pragma unroll
      for (int j = 0; j < 16; ++j) {
        float x = c_sh[wave * HH + lane + j * 32];
        s += x;
        q += x * x;
      }
#pragma unroll
      for (int off = 16; off > 0; off >>= 1) {
        s += __shfl_down(s, off, 32);
        q += __shfl_down(q, off, 32);
      }
      if (lane == 0) { s_cs[wave] = s; s_cq[wave] = q; }
    }
    __syncthreads();

    // ---- h_new; write outputs; refresh LDS-resident h ----
#pragma unroll
    for (int r = 0; r < 16; ++r) {
      float mu   = s_cs[r] * (1.0f / HH);
      float var  = s_cq[r] * (1.0f / HH) - mu * mu;
      float rstd = rsqrtf(var + 1e-5f);
      float lnc  = (cn[r] - mu) * rstd * gmh + bth;
      float ov   = v_sh[r * VSTR + hh + 1536];
      float hn   = sigf(ov) * tanhf(lnc);
      size_t ob  = (size_t)(t + 1) * BB * HH + (size_t)(b0 + r) * HH + hh;
      h_base[ob] = hn;
      c_base[ob] = cn[r];
      h_sh[r * HPAD + hh] = f2bf(hn);
    }
    // loop-top __syncthreads() orders h_sh for the next step's GEMM
  }
}

// ------------------------------- launcher ----------------------------------

extern "C" void kernel_launch(void* const* d_in, const int* in_sizes, int n_in,
                              void* d_out, int out_size, void* d_ws, size_t ws_size,
                              hipStream_t stream) {
  const float* x       = (const float*)d_in[0];   // [T,B,I]
  const float* h0      = (const float*)d_in[1];   // [B,H]
  const float* c0      = (const float*)d_in[2];   // [B,H]
  const float* W       = (const float*)d_in[3];   // [I,4H]
  const float* R       = (const float*)d_in[4];   // [H,4H]
  const float* bias    = (const float*)d_in[5];   // [4H]
  const float* gamma   = (const float*)d_in[6];   // [2,4H]
  const float* gamma_h = (const float*)d_in[7];   // [H]
  const float* beta_h  = (const float*)d_in[8];   // [H]

  float* out = (float*)d_out;
  const size_t HC = (size_t)(TT + 1) * BB * HH;
  float* h_base = out;        // [T+1,B,H]
  float* c_base = out + HC;   // [T+1,B,H]

  // workspace carve-up (region sizes are 256B multiples)
  char* w = (char*)d_ws;
  size_t off = 0;
  float* Wx = (float*)(w + off);                      off += (size_t)TT * BB * GG * 4;
  unsigned short* x_bf = (unsigned short*)(w + off);  off += (size_t)TT * BB * II * 2;
  unsigned short* Wp   = (unsigned short*)(w + off);  off += (size_t)II * GG * 2;
  unsigned short* Rp   = (unsigned short*)(w + off);  off += (size_t)HH * GG * 2;
  (void)ws_size; (void)n_in; (void)in_sizes; (void)out_size;

  const float* gamma0 = gamma;
  const float* gamma1 = gamma + GG;

  // phase 0: converts / weight packing / state init
  {
    int n = TT * BB * II;
    f32_to_bf16_kernel<<<(n + 255) / 256, 256, 0, stream>>>(x, x_bf, n);
  }
  {
    int total = (KD / 32) * (GG / 16) * 32;
    pack_b_kernel<<<(total + 255) / 256, 256, 0, stream>>>(W, Wp, KD, GG);
    pack_b_kernel<<<(total + 255) / 256, 256, 0, stream>>>(R, Rp, KD, GG);
  }
  init_state_kernel<<<(BB * HH + 255) / 256, 256, 0, stream>>>(
      h0, c0, h_base, c_base, BB * HH);

  // phase 1: Wx = LN(x @ W, gamma0) for all timesteps
  gemm_ln_kernel<<<(TT * BB) / 16, 256, 0, stream>>>(x_bf, Wp, gamma0, Wx);

  // phase 2: one persistent kernel runs the whole recurrence
  const size_t lds_bytes = (size_t)16 * VSTR * 4     // v_sh (padded)
                         + (size_t)16 * HH * 4       // c_sh
                         + (size_t)16 * HPAD * 2;    // h_sh (padded bf16)
  lstm_persistent_kernel<<<BB / 16, 512, lds_bytes, stream>>>(
      Rp, Wx, bias, gamma1, gamma_h, beta_h, h0, c0, h_base, c_base);
}
